// CUDAGraphMoEExperts_81527069212684
// MI455X (gfx1250) — compile-verified, matrix operand order checked by reading, not run on hardware
//
#include <hip/hip_runtime.h>
#include <hip/hip_bf16.h>
#include <math.h>

// Problem sizes (fixed by the reference)
#define E_ 8
#define H_ 2048
#define I_ 4096
#define B_ 2048
#define K_ 2
#define P_ (B_ * K_)          // 4096 routed (token, expert) pairs total

// GEMM tiling: 256 threads = 8 waves; macro tile 128(M) x 64(N), K-step 32.
// Waves arranged 4(M) x 2(N); each wave owns a 32x32 output tile (2x2 WMMA frags).
#define BM 128
#define BN 64
#define BK 32
#define LDA 40                 // LDS row stride in bf16 elements (80 B, 16B-aligned)

typedef __attribute__((ext_vector_type(16))) __bf16 v16bf;
typedef __attribute__((ext_vector_type(8)))  float  v8f;

union Frag { unsigned u[8]; v16bf v; };

#define HAS_ASYNC_LDS __has_builtin(__builtin_amdgcn_global_load_async_to_lds_b128)
#if HAS_ASYNC_LDS
typedef __attribute__((ext_vector_type(4))) int v4i_t;
typedef __attribute__((address_space(1))) v4i_t g_v4i;   // global int4
typedef __attribute__((address_space(3))) v4i_t l_v4i;   // LDS int4
#endif

__device__ __forceinline__ void wait_async_lds() {
#if HAS_ASYNC_LDS
#if __has_builtin(__builtin_amdgcn_s_wait_asynccnt)
  __builtin_amdgcn_s_wait_asynccnt(0);
#else
  asm volatile("s_wait_asynccnt 0x0" ::: "memory");
#endif
#endif
}

__device__ __forceinline__ unsigned short f2bf(float f) {
  unsigned u = __float_as_uint(f);
  u += 0x7FFFu + ((u >> 16) & 1u);       // round-to-nearest-even
  return (unsigned short)(u >> 16);
}

// Load one 16x32 bf16 operand fragment from LDS (ISA 7.12.2 layout):
// lane%16 selects the row (M for A, N for B); VGPR v holds K pair
// k = (v/4)*16 + (lane/16)*8 + (v%4)*2  -> 16B-contiguous per 4 VGPRs,
// so the compiler lowers this to 2x ds_load_b128 per fragment.
__device__ __forceinline__ void load_frag(const unsigned short* __restrict__ s,
                                          int lane, Frag& f) {
  const int half8 = (lane >> 4) << 3;               // 0 or 8
  const unsigned short* row = s + (lane & 15) * LDA;
  #pragma unroll
  for (int v = 0; v < 8; ++v) {
    const int k = ((v >> 2) << 4) + half8 + ((v & 3) << 1);
    f.u[v] = *(const unsigned*)(row + k);           // two bf16 per dword
  }
}

// ---------------- routing ----------------

__global__ void moe_count(const int* __restrict__ idx, int* __restrict__ counts) {
  int i = blockIdx.x * 256 + threadIdx.x;
  if (i < P_) {
    int e = idx[i] & (E_ - 1);
    atomicAdd(&counts[e], 1);
  }
}

__global__ void moe_scan(const int* __restrict__ counts,
                         int* __restrict__ offsets, int* __restrict__ cursor) {
  if (threadIdx.x == 0) {
    int acc = 0;
    for (int e = 0; e < E_; ++e) {
      offsets[e] = acc;
      cursor[e]  = acc;
      acc += counts[e];
    }
    offsets[E_] = acc;
  }
}

__global__ void moe_scatter(const int* __restrict__ idx, int* __restrict__ cursor,
                            int* __restrict__ tok_of_slot, int* __restrict__ slot_of_bk) {
  int i = blockIdx.x * 256 + threadIdx.x;
  if (i < P_) {
    int e = idx[i] & (E_ - 1);
    int s = atomicAdd(&cursor[e], 1);
    tok_of_slot[s] = i >> 1;          // token id b  (K_ == 2)
    slot_of_bk[i]  = s;
  }
}

// ---------------- GEMM1: act = silu(x@w1) * (x@w3), grouped per expert ----------------

__global__ __launch_bounds__(256, 2) void moe_gemm1(
    const float* __restrict__ x, const float* __restrict__ w1, const float* __restrict__ w3,
    const int* __restrict__ counts, const int* __restrict__ offsets,
    const int* __restrict__ tok_of_slot, unsigned short* __restrict__ act) {
  const int e = blockIdx.z;
  const int cnt = counts[e];
  const int m_base = blockIdx.y * BM;
  if (m_base >= cnt) return;                       // uniform early exit
  const int off = offsets[e];
  const int n0 = blockIdx.x * BN;
  const int tid = threadIdx.x;
  const int lane = tid & 31, wave = tid >> 5;
  const int wm = wave & 3, wn = wave >> 2;

  __shared__ unsigned short As[BM * LDA];          // x tile, bf16 [m][k]
  __shared__ unsigned short B1s[BN * LDA];         // w1 tile, bf16 [n][k]
  __shared__ unsigned short B3s[BN * LDA];         // w3 tile, bf16 [n][k]

  // Each thread stages 4 rows of A; resolve their token ids once.
  const int rr = tid >> 3;                          // 0..31
  const int cg = (tid & 7) * 4;                     // 0..28
  int tok[4];
  #pragma unroll
  for (int j = 0; j < 4; ++j) {
    int pos = m_base + rr + 32 * j;
    tok[j] = (pos < cnt) ? tok_of_slot[off + pos] : 0;
  }

  v8f accg[2][2] = {};
  v8f accu[2][2] = {};

  #pragma unroll 1
  for (int k0 = 0; k0 < H_; k0 += BK) {
    // A: gather x rows, fp32 -> bf16, packed dword stores (ds_store_b64)
    #pragma unroll
    for (int j = 0; j < 4; ++j) {
      const float4 v = *(const float4*)(x + (size_t)tok[j] * H_ + k0 + cg);
      uint2 pk;
      pk.x = (unsigned)f2bf(v.x) | ((unsigned)f2bf(v.y) << 16);
      pk.y = (unsigned)f2bf(v.z) | ((unsigned)f2bf(v.w) << 16);
      *(uint2*)&As[(rr + 32 * j) * LDA + cg] = pk;
    }
    // B: stream w1/w3 fp32, convert, transpose to [n][k]
    #pragma unroll
    for (int j = 0; j < 2; ++j) {
      int lin = tid + 256 * j;                      // 0..511
      int k = lin >> 4;                             // 0..31
      int n = (lin & 15) * 4;                       // 0..60
      size_t goff = ((size_t)e * H_ + (size_t)(k0 + k)) * I_ + n0 + n;
      float4 a = *(const float4*)(w1 + goff);
      float4 b = *(const float4*)(w3 + goff);
      B1s[(n + 0) * LDA + k] = f2bf(a.x); B1s[(n + 1) * LDA + k] = f2bf(a.y);
      B1s[(n + 2) * LDA + k] = f2bf(a.z); B1s[(n + 3) * LDA + k] = f2bf(a.w);
      B3s[(n + 0) * LDA + k] = f2bf(b.x); B3s[(n + 1) * LDA + k] = f2bf(b.y);
      B3s[(n + 2) * LDA + k] = f2bf(b.z); B3s[(n + 3) * LDA + k] = f2bf(b.w);
    }
    __syncthreads();

    Frag a[2], b1[2], b3[2];
    #pragma unroll
    for (int fm = 0; fm < 2; ++fm)
      load_frag(&As[(wm * 32 + fm * 16) * LDA], lane, a[fm]);
    #pragma unroll
    for (int fn = 0; fn < 2; ++fn) {
      load_frag(&B1s[(wn * 32 + fn * 16) * LDA], lane, b1[fn]);
      load_frag(&B3s[(wn * 32 + fn * 16) * LDA], lane, b3[fn]);
    }
    #pragma unroll
    for (int fm = 0; fm < 2; ++fm) {
      #pragma unroll
      for (int fn = 0; fn < 2; ++fn) {
        accg[fm][fn] = __builtin_amdgcn_wmma_f32_16x16x32_bf16(
            false, a[fm].v, false, b1[fn].v, (short)0, accg[fm][fn], false, false);
        accu[fm][fn] = __builtin_amdgcn_wmma_f32_16x16x32_bf16(
            false, a[fm].v, false, b3[fn].v, (short)0, accu[fm][fn], false, false);
      }
    }
    __syncthreads();
  }

  // epilogue: act = silu(g) * u, bf16, predicated on row validity.
  // sigmoid via v_exp_f32 + v_rcp_f32 (no IEEE divide expansion).
  const int half8 = (lane >> 4) << 3;
  const int nl = lane & 15;
  #pragma unroll
  for (int fm = 0; fm < 2; ++fm) {
    #pragma unroll
    for (int fn = 0; fn < 2; ++fn) {
      #pragma unroll
      for (int r = 0; r < 8; ++r) {
        int m = wm * 32 + fm * 16 + half8 + r;
        int pos = m_base + m;
        if (pos < cnt) {
          float g = accg[fm][fn][r];
          float u = accu[fm][fn][r];
          float sig = __builtin_amdgcn_rcpf(1.f + __expf(-g));
          float s = g * sig * u;                    // silu(g) * u
          int n = n0 + wn * 32 + fn * 16 + nl;
          act[(size_t)(off + pos) * I_ + n] = f2bf(s);
        }
      }
    }
  }
}

// ---------------- GEMM2: out_pairs = act @ w2, grouped per expert ----------------

__global__ __launch_bounds__(256, 2) void moe_gemm2(
    const unsigned short* __restrict__ act, const float* __restrict__ w2,
    const int* __restrict__ counts, const int* __restrict__ offsets,
    float* __restrict__ out_pairs) {
  const int e = blockIdx.z;
  const int cnt = counts[e];
  const int m_base = blockIdx.y * BM;
  if (m_base >= cnt) return;
  const int off = offsets[e];
  const int n0 = blockIdx.x * BN;
  const int tid = threadIdx.x;
  const int lane = tid & 31, wave = tid >> 5;
  const int wm = wave & 3, wn = wave >> 2;

  __shared__ unsigned short As[BM * LDA];
  __shared__ unsigned short Bs[BN * LDA];

  v8f acc[2][2] = {};

  #pragma unroll 1
  for (int k0 = 0; k0 < I_; k0 += BK) {
    // A: act rows (already bf16), 128x32 per step -- pure copy, use the
    // CDNA5 async global->LDS path (GLOBAL_LOAD_ASYNC_TO_LDS_B128).
    #pragma unroll
    for (int j = 0; j < 2; ++j) {
      int lin = tid + 256 * j;                      // 0..511
      int row = lin >> 2;                           // 0..127
      int grp = lin & 3;                            // 8 bf16 each
      int slot = off + m_base + row;
      if (slot >= P_) slot = P_ - 1;                // clamp (tail rows unused)
      const unsigned short* gsrc = act + (size_t)slot * I_ + k0 + grp * 8;
      unsigned short* ldst = &As[row * LDA + grp * 8];
#if HAS_ASYNC_LDS
      __builtin_amdgcn_global_load_async_to_lds_b128(
          (g_v4i*)gsrc, (l_v4i*)ldst, 0, 0);
#else
      *(uint4*)ldst = *(const uint4*)gsrc;
#endif
    }
    // B: w2 fp32 -> bf16, transpose to [n][k]
    #pragma unroll
    for (int j = 0; j < 2; ++j) {
      int lin = tid + 256 * j;
      int k = lin >> 4;
      int n = (lin & 15) * 4;
      size_t goff = ((size_t)e * I_ + (size_t)(k0 + k)) * H_ + n0 + n;
      float4 a = *(const float4*)(w2 + goff);
      Bs[(n + 0) * LDA + k] = f2bf(a.x); Bs[(n + 1) * LDA + k] = f2bf(a.y);
      Bs[(n + 2) * LDA + k] = f2bf(a.z); Bs[(n + 3) * LDA + k] = f2bf(a.w);
    }
    wait_async_lds();
    __syncthreads();

    Frag a[2], b[2];
    #pragma unroll
    for (int fm = 0; fm < 2; ++fm)
      load_frag(&As[(wm * 32 + fm * 16) * LDA], lane, a[fm]);
    #pragma unroll
    for (int fn = 0; fn < 2; ++fn)
      load_frag(&Bs[(wn * 32 + fn * 16) * LDA], lane, b[fn]);
    #pragma unroll
    for (int fm = 0; fm < 2; ++fm) {
      #pragma unroll
      for (int fn = 0; fn < 2; ++fn) {
        acc[fm][fn] = __builtin_amdgcn_wmma_f32_16x16x32_bf16(
            false, a[fm].v, false, b[fn].v, (short)0, acc[fm][fn], false, false);
      }
    }
    __syncthreads();
  }

  const int half8 = (lane >> 4) << 3;
  const int nl = lane & 15;
  #pragma unroll
  for (int fm = 0; fm < 2; ++fm) {
    #pragma unroll
    for (int fn = 0; fn < 2; ++fn) {
      #pragma unroll
      for (int r = 0; r < 8; ++r) {
        int m = wm * 32 + fm * 16 + half8 + r;
        int pos = m_base + m;
        if (pos < cnt) {
          int n = n0 + wn * 32 + fn * 16 + nl;
          out_pairs[(size_t)(off + pos) * H_ + n] = acc[fm][fn][r];
        }
      }
    }
  }
}

// ---------------- combine: out[b] = sum_k ew[b,k] * out_pairs[slot(b,k)] ----------------

__global__ void moe_combine(const float* __restrict__ out_pairs,
                            const int* __restrict__ slot_of_bk,
                            const float* __restrict__ ew, float* __restrict__ out) {
  int b = blockIdx.x;
  int h = blockIdx.y * 256 + threadIdx.x;
  int s0 = slot_of_bk[b * K_ + 0];
  int s1 = slot_of_bk[b * K_ + 1];
  float e0 = ew[b * K_ + 0];
  float e1 = ew[b * K_ + 1];
  out[(size_t)b * H_ + h] =
      e0 * out_pairs[(size_t)s0 * H_ + h] + e1 * out_pairs[(size_t)s1 * H_ + h];
}

// ---------------- launch ----------------

extern "C" void kernel_launch(void* const* d_in, const int* in_sizes, int n_in,
                              void* d_out, int out_size, void* d_ws, size_t ws_size,
                              hipStream_t stream) {
  const float* x  = (const float*)d_in[0];
  const int* idx  = (const int*)d_in[1];      // expert_indices
  const float* ew = (const float*)d_in[2];    // expert_weights
  const float* w1 = (const float*)d_in[3];
  const float* w2 = (const float*)d_in[4];
  const float* w3 = (const float*)d_in[5];
  float* out = (float*)d_out;

  char* ws = (char*)d_ws;
  int* counts      = (int*)ws;                       // E_
  int* offsets     = counts + E_;                    // E_+1
  int* cursor      = offsets + E_ + 1;               // E_
  int* tok_of_slot = (int*)(ws + 1024);              // P_
  int* slot_of_bk  = tok_of_slot + P_;               // P_
  unsigned short* act = (unsigned short*)(ws + 64 * 1024);                   // P_*I_ bf16 (33.5 MB)
  float* out_pairs = (float*)(ws + 64 * 1024 + (size_t)P_ * I_ * 2);         // P_*H_ f32 (33.5 MB)

  (void)hipMemsetAsync(counts, 0, E_ * sizeof(int), stream);
  moe_count  <<<dim3((P_ + 255) / 256), dim3(256), 0, stream>>>(idx, counts);
  moe_scan   <<<dim3(1), dim3(32), 0, stream>>>(counts, offsets, cursor);
  moe_scatter<<<dim3((P_ + 255) / 256), dim3(256), 0, stream>>>(idx, cursor, tok_of_slot, slot_of_bk);
  moe_gemm1  <<<dim3(I_ / BN, P_ / BM, E_), dim3(256), 0, stream>>>(x, w1, w3, counts, offsets, tok_of_slot, act);
  moe_gemm2  <<<dim3(H_ / BN, P_ / BM, E_), dim3(256), 0, stream>>>(act, w2, counts, offsets, out_pairs);
  moe_combine<<<dim3(B_, H_ / 256), dim3(256), 0, stream>>>(out_pairs, slot_of_bk, ew, out);

  (void)in_sizes; (void)n_in; (void)out_size; (void)ws_size;
}
